// Encoder_78271484002491
// MI455X (gfx1250) — compile-verified, matrix operand order checked by reading, not run on hardware
//
#include <hip/hip_runtime.h>
#include <hip/hip_bf16.h>

// ---------------- problem constants ----------------
constexpr int Bc = 16, Tc = 128, Dc = 41, NKc = 128, Hc = 4, NLc = 3, Lc = 2624;
constexpr int TD = Tc * Dc;             // 5248
constexpr int BL = Bc * Lc;             // 41984

typedef _Float16 half_t;
typedef __attribute__((ext_vector_type(16))) _Float16 v16h;
typedef __attribute__((ext_vector_type(8)))  float    v8f;

__device__ __forceinline__ v8f wmma_f16(v16h a, v16h b, v8f c) {
    return __builtin_amdgcn_wmma_f32_16x16x32_f16(false, a, false, b, (short)0, c,
                                                  false, false);
}

// ---------------- pack (stable compaction of mask==1) ----------------
__global__ __launch_bounds__(256) void pack_kernel(
    const float* __restrict__ mask, const float* __restrict__ tmask,
    const float* __restrict__ value,
    int* __restrict__ T_inds, int* __restrict__ C_inds, float* __restrict__ Uvals,
    float* __restrict__ out_tgt, float* __restrict__ out_src,
    float* __restrict__ out_smask)
{
    const int b = blockIdx.x;
    const int tid = threadIdx.x;
    __shared__ int cnt[256];
    __shared__ int offs[256];
    const int per = (TD + 255) / 256;          // 21
    int start = tid * per;
    int end = start + per; if (end > TD) end = TD;
    const float* mb = mask + (size_t)b * TD;
    int c = 0;
    for (int i = start; i < end; ++i) c += (mb[i] != 0.f);
    cnt[tid] = c;
    __syncthreads();
    if (tid == 0) {
        int s = 0;
        for (int i = 0; i < 256; ++i) { offs[i] = s; s += cnt[i]; }
    }
    __syncthreads();
    int p = offs[tid];
    for (int i = start; i < end; ++i) {
        if (mb[i] != 0.f && p < Lc) {
            const size_t o = (size_t)b * Lc + p;
            T_inds[o] = i / Dc;
            C_inds[o] = i % Dc;
            const float v = value[(size_t)b * TD + i];
            const float tg = tmask[(size_t)b * TD + i];
            Uvals[o] = v;
            out_tgt[o] = tg;
            out_src[o] = v;
            out_smask[o] = 1.f - tg;
            ++p;
        }
    }
}

// ---------------- corr normalize ----------------
__global__ __launch_bounds__(256) void corr_kernel(
    const float* __restrict__ corr, const int* __restrict__ i_patch,
    float* __restrict__ cpn)
{
    __shared__ float smin[256], smax[256];
    const int tid = threadIdx.x;
    const int ip = i_patch[0];
    const float* c = corr + (size_t)ip * Dc * Dc;
    float mn = 3.0e38f, mx = -3.0e38f;
    for (int i = tid; i < Dc * Dc; i += 256) {
        const int r = i / Dc, cc = i % Dc;
        const float v = 0.5f * (c[i] + c[cc * Dc + r]);
        mn = fminf(mn, v); mx = fmaxf(mx, v);
    }
    smin[tid] = mn; smax[tid] = mx;
    __syncthreads();
    for (int s = 128; s > 0; s >>= 1) {
        if (tid < s) {
            smin[tid] = fminf(smin[tid], smin[tid + s]);
            smax[tid] = fmaxf(smax[tid], smax[tid + s]);
        }
        __syncthreads();
    }
    const float lo = smin[0];
    const float inv = 1.f / (smax[0] - lo + 1e-6f);
    for (int i = tid; i < Dc * Dc; i += 256) {
        const int r = i / Dc, cc = i % Dc;
        const float v = 0.5f * (c[i] + c[cc * Dc + r]);
        cpn[i] = (v - lo) * inv;
    }
}

// ---------------- channel embedding (tiny GEMM, broadcast over batch) ----------------
__global__ __launch_bounds__(128) void chan_embed_kernel(
    const float* __restrict__ cpn, const float* __restrict__ w,
    const float* __restrict__ bias, float* __restrict__ Cemb)
{
    const int d = blockIdx.x, n = threadIdx.x;
    float acc = bias[n];
    for (int k = 0; k < Dc; ++k) acc += cpn[d * Dc + k] * w[n * Dc + k];
    const float v = fmaxf(acc, 0.f);
    for (int b = 0; b < Bc; ++b)
        Cemb[((size_t)b * Dc + d) * NKc + n] = v;
}

// ---------------- time embedding ----------------
__global__ __launch_bounds__(256) void time_embed_kernel(
    const float* __restrict__ ctx, const float* __restrict__ tw,
    const float* __restrict__ tb, float* __restrict__ Temb)
{
    const int idx = blockIdx.x * 256 + threadIdx.x;
    if (idx >= Bc * Tc * NKc) return;
    const int n = idx % NKc;
    const int bt = idx / NKc;
    Temb[idx] = sinf(ctx[bt] * tw[n] + tb[n]);
}

// ---------------- edge embedding ----------------
__global__ __launch_bounds__(256) void edge_embed_kernel(
    const float* __restrict__ Uvals, const float* __restrict__ tgt,
    const float* __restrict__ ew, const float* __restrict__ eb,
    float* __restrict__ U)
{
    const int idx = blockIdx.x * 256 + threadIdx.x;
    if (idx >= BL * NKc) return;
    const int n = idx % NKc;
    const int bl = idx / NKc;
    const float uv = Uvals[bl];
    const float ui = tgt[bl];        // U_ind = 1-mk+tgt = tgt (mk == 1)
    U[idx] = fmaxf(uv * ew[n * 2 + 0] + ui * ew[n * 2 + 1] + eb[n], 0.f);
}

// ---------------- concat [gather(emb, inds), U] -> [B, L, 256] ----------------
__global__ __launch_bounds__(256) void cat2_kernel(
    const float* __restrict__ emb, int rowsPerB, const int* __restrict__ inds,
    const float* __restrict__ U, float* __restrict__ cat2)
{
    const int idx = blockIdx.x * 256 + threadIdx.x;
    if (idx >= BL * 2 * NKc) return;
    const int n = idx % (2 * NKc);
    const size_t bl = idx / (2 * NKc);
    const int b = (int)(bl / Lc);
    float v;
    if (n < NKc) {
        const int r = inds[bl];
        v = emb[((size_t)b * rowsPerB + r) * NKc + n];
    } else {
        v = U[bl * NKc + (n - NKc)];
    }
    cat2[idx] = v;
}

// ---------------- concat [U, gather(Temb), gather(Cemb)] -> [B, L, 384] ----------------
__global__ __launch_bounds__(256) void cat3_kernel(
    const float* __restrict__ U, const float* __restrict__ Temb,
    const int* __restrict__ T_inds, const float* __restrict__ Cemb,
    const int* __restrict__ C_inds, float* __restrict__ cat3)
{
    const int idx = blockIdx.x * 256 + threadIdx.x;
    if (idx >= BL * 3 * NKc) return;
    const int n = idx % (3 * NKc);
    const size_t bl = idx / (3 * NKc);
    const int b = (int)(bl / Lc);
    float v;
    if (n < NKc) {
        v = U[bl * NKc + n];
    } else if (n < 2 * NKc) {
        const int t = T_inds[bl];
        v = Temb[((size_t)b * Tc + t) * NKc + (n - NKc)];
    } else {
        const int c = C_inds[bl];
        v = Cemb[((size_t)b * Dc + c) * NKc + (n - 2 * NKc)];
    }
    cat3[idx] = v;
}

// ---------------- generic WMMA GEMM: Y = epilogue(X @ W^T + bias) ----------------
// X [M, Kd] f32, W [N, Kd] f32, Y [M, N] f32, N multiple of 64, Kd multiple of 32.
// mode 0: acc+b  | 1: relu(acc+b) | 2: relu(R+acc+b) | 3: R+relu(acc+b)
__global__ __launch_bounds__(256) void gemm_kernel(
    const float* __restrict__ X, const float* __restrict__ W,
    const float* __restrict__ bias, const float* __restrict__ R,
    float* __restrict__ Y, int M, int Kd, int N, int mode)
{
    __shared__ half_t As[64][40];
    __shared__ half_t Bs[64][40];
    const int tid = threadIdx.x;
    const int lane = tid & 31, wv = tid >> 5;
    const int mblk = blockIdx.x * 64, nblk = blockIdx.y * 64;
    const int wm = (wv & 3) * 16;     // wave's row offset in tile
    const int wn = (wv >> 2) * 32;    // wave's col offset in tile
    const int lr = tid >> 2;          // loader row 0..63
    const int lk = (tid & 3) * 8;     // loader k chunk
    const int l16 = lane & 15;
    const int koff = (lane >> 4) * 8;
    v8f c0 = {}, c1 = {};
    const int gm = mblk + lr;
    const int gmc = (gm < M) ? gm : (M - 1);     // clamped row: load unconditionally
    const half_t hz = (half_t)0.f;
    const bool mvalid = (gm < M);
    for (int kb = 0; kb < Kd; kb += 32) {
        const float* xp = X + (size_t)gmc * Kd + kb + lk;
        const float* wp = W + (size_t)(nblk + lr) * Kd + kb + lk;
        if (kb + 32 < Kd) {                       // WGP-scope prefetch of next k-tile
            __builtin_prefetch(xp + 32, 0, 3);
            __builtin_prefetch(wp + 32, 0, 3);
        }
        #pragma unroll
        for (int i = 0; i < 8; ++i) {
            const float xv = xp[i];               // always in-bounds (clamped)
            As[lr][lk + i] = mvalid ? (half_t)xv : hz;
            Bs[lr][lk + i] = (half_t)wp[i];
        }
        __syncthreads();
        v16h a, b0, b1;
        const int ar = wm + l16;
        const int bc0 = wn + l16, bc1 = bc0 + 16;
        #pragma unroll
        for (int i = 0; i < 8; ++i) {
            a[i]      = As[ar][koff + i];
            a[i + 8]  = As[ar][16 + koff + i];
            b0[i]     = Bs[bc0][koff + i];
            b0[i + 8] = Bs[bc0][16 + koff + i];
            b1[i]     = Bs[bc1][koff + i];
            b1[i + 8] = Bs[bc1][16 + koff + i];
        }
        c0 = wmma_f16(a, b0, c0);
        c1 = wmma_f16(a, b1, c1);
        __syncthreads();
    }
    const int mlo = mblk + wm + (lane >> 4) * 8;
    const int n0 = nblk + wn + l16;
    const int n1 = n0 + 16;
    const float bia0 = bias ? bias[n0] : 0.f;
    const float bia1 = bias ? bias[n1] : 0.f;
    #pragma unroll
    for (int r = 0; r < 8; ++r) {
        const int m = mlo + r;
        if (m >= M) continue;
        float v0 = c0[r] + bia0, v1 = c1[r] + bia1;
        if (mode == 1) {
            v0 = fmaxf(v0, 0.f); v1 = fmaxf(v1, 0.f);
        } else if (mode == 2) {
            v0 = fmaxf(R[(size_t)m * N + n0] + v0, 0.f);
            v1 = fmaxf(R[(size_t)m * N + n1] + v1, 0.f);
        } else if (mode == 3) {
            v0 = R[(size_t)m * N + n0] + fmaxf(v0, 0.f);
            v1 = R[(size_t)m * N + n1] + fmaxf(v1, 0.f);
        }
        Y[(size_t)m * N + n0] = v0;
        Y[(size_t)m * N + n1] = v1;
    }
}

// ---------------- flash attention (per-wave q-tile, H=4 heads of d=32) ----------------
// AO[b,q,:] = Qp[b,q,:] + softmax(Qp Kp^T / sqrt(128), mask) @ Vp
// maskmode 0: none; 1: valid iff inds[b,k]==q  (score else -1e10, as reference)
// All fragment loads use CLAMPED row indices + unconditional loads so the v16h
// fragments stay in VGPRs (no exec-divergent element inserts). Padded score
// columns are overwritten with -1e30 after the WMMA; padded V rows multiply
// P = exp(-1e30 - m) = 0, so clamped data never affects stored results.
__global__ __launch_bounds__(256) void attn_kernel(
    const float* __restrict__ Qp, const float* __restrict__ Kp,
    const float* __restrict__ Vp, const int* __restrict__ inds,
    float* __restrict__ AO, int nq, int nk, int maskmode)
{
    __shared__ half_t Pbuf[8][16][40];
    const int tid = threadIdx.x, lane = tid & 31, wv = tid >> 5;
    const int QT = (nq + 15) >> 4;
    const int gw = blockIdx.x * 8 + wv;
    if (gw >= Bc * Hc * QT) return;                       // wave-uniform exit
    const int qt = gw % QT;
    const int h  = (gw / QT) % Hc;
    const int b  = gw / (QT * Hc);
    const int mbase = qt * 16;
    const int hd = h * 32;
    const float* Qb = Qp + (size_t)b * nq * NKc;
    const float* Kb = Kp + (size_t)b * nk * NKc;
    const float* Vb = Vp + (size_t)b * nk * NKc;
    const int* indb = maskmode ? (inds + (size_t)b * nk) : nullptr;
    const int l16 = lane & 15;
    const int koff = (lane >> 4) * 8;
    const int myrow0 = mbase + (lane >> 4) * 8;

    v16h aq;
    {
        const int qrow = mbase + l16;
        const int qrc = (qrow < nq) ? qrow : (nq - 1);    // clamp, load always
        const float* qr = Qb + (size_t)qrc * NKc + hd;
        #pragma unroll
        for (int i = 0; i < 8; ++i) {
            aq[i]     = (half_t)qr[koff + i];
            aq[i + 8] = (half_t)qr[16 + koff + i];
        }
    }
    const float scale = 0.0883883476483184f;   // 1/sqrt(128)
    float mrow[8], lrow[8];
    #pragma unroll
    for (int r = 0; r < 8; ++r) { mrow[r] = -3.0e38f; lrow[r] = 0.f; }
    v8f o0 = {}, o1 = {};

    for (int kt = 0; kt < nk; kt += 32) {
        const int kc0 = kt + l16, kc1 = kt + 16 + l16;
        const int kr0 = (kc0 < nk) ? kc0 : 0;             // clamped rows
        const int kr1 = (kc1 < nk) ? kc1 : 0;
        v16h bk0, bk1;
        {
            const float* kra = Kb + (size_t)kr0 * NKc + hd;
            const float* krb = Kb + (size_t)kr1 * NKc + hd;
            #pragma unroll
            for (int i = 0; i < 8; ++i) {
                bk0[i]     = (half_t)kra[koff + i];
                bk0[i + 8] = (half_t)kra[16 + koff + i];
                bk1[i]     = (half_t)krb[koff + i];
                bk1[i + 8] = (half_t)krb[16 + koff + i];
            }
        }
        v8f s0 = {}, s1 = {};
        s0 = wmma_f16(aq, bk0, s0);
        s1 = wmma_f16(aq, bk1, s1);

        int ind0 = -1, ind1 = -1;
        if (maskmode) {
            ind0 = indb[kr0];                              // clamped (row 0 dummy)
            ind1 = indb[kr1];
        }
        float p0[8], p1[8];
        #pragma unroll
        for (int r = 0; r < 8; ++r) {
            float v0 = s0[r] * scale, v1 = s1[r] * scale;
            if (maskmode) {
                const int q = myrow0 + r;
                v0 = (ind0 == q) ? v0 : -1e10f;
                v1 = (ind1 == q) ? v1 : -1e10f;
            }
            if (kc0 >= nk) v0 = -1e30f;                    // padding kill (assign, so
            if (kc1 >= nk) v1 = -1e30f;                    //  clamped garbage vanishes)
            p0[r] = v0; p1[r] = v1;
        }
        #pragma unroll
        for (int r = 0; r < 8; ++r) {
            float t = fmaxf(p0[r], p1[r]);
            t = fmaxf(t, __shfl_xor(t, 1, 32));
            t = fmaxf(t, __shfl_xor(t, 2, 32));
            t = fmaxf(t, __shfl_xor(t, 4, 32));
            t = fmaxf(t, __shfl_xor(t, 8, 32));
            const float nm = fmaxf(mrow[r], t);
            const float corr = __expf(mrow[r] - nm);
            mrow[r] = nm;
            const float e0 = __expf(p0[r] - nm);
            const float e1 = __expf(p1[r] - nm);
            p0[r] = e0; p1[r] = e1;
            float ps = e0 + e1;
            ps += __shfl_xor(ps, 1, 32);
            ps += __shfl_xor(ps, 2, 32);
            ps += __shfl_xor(ps, 4, 32);
            ps += __shfl_xor(ps, 8, 32);
            lrow[r] = lrow[r] * corr + ps;
            o0[r] *= corr; o1[r] *= corr;
        }
        // transpose P tile through per-wave LDS (C-layout -> A-fragment layout)
        #pragma unroll
        for (int r = 0; r < 8; ++r) {
            const int mr = (lane >> 4) * 8 + r;
            Pbuf[wv][mr][l16]      = (half_t)p0[r];
            Pbuf[wv][mr][16 + l16] = (half_t)p1[r];
        }
        __builtin_amdgcn_wave_barrier();
        v16h pa, bv0, bv1;
        #pragma unroll
        for (int i = 0; i < 8; ++i) {
            pa[i]     = Pbuf[wv][l16][koff + i];
            pa[i + 8] = Pbuf[wv][l16][16 + koff + i];
        }
        __builtin_amdgcn_wave_barrier();
        #pragma unroll
        for (int i = 0; i < 8; ++i) {
            const int k0 = kt + koff + i;
            const int k1 = kt + 16 + koff + i;
            const int k0c = (k0 < nk) ? k0 : 0;            // clamp; P rows there are 0
            const int k1c = (k1 < nk) ? k1 : 0;
            const float* vra = Vb + (size_t)k0c * NKc + hd;
            const float* vrb = Vb + (size_t)k1c * NKc + hd;
            bv0[i]     = (half_t)vra[l16];
            bv1[i]     = (half_t)vra[16 + l16];
            bv0[i + 8] = (half_t)vrb[l16];
            bv1[i + 8] = (half_t)vrb[16 + l16];
        }
        o0 = wmma_f16(pa, bv0, o0);
        o1 = wmma_f16(pa, bv1, o1);
    }
    #pragma unroll
    for (int r = 0; r < 8; ++r) {
        const int m = myrow0 + r;
        if (m >= nq) continue;
        const float inv = 1.f / lrow[r];
        const size_t base = ((size_t)b * nq + m) * NKc + hd;
        AO[base + l16]      = Qb[(size_t)m * NKc + hd + l16]      + o0[r] * inv;
        AO[base + 16 + l16] = Qb[(size_t)m * NKc + hd + 16 + l16] + o1[r] * inv;
    }
}

// ---------------- final out projection: [B*L, 384] @ [384] + b -> [B*L] ----------------
__global__ __launch_bounds__(256) void outproj_kernel(
    const float* __restrict__ cat3, const float* __restrict__ ow,
    const float* __restrict__ ob, float* __restrict__ out)
{
    const int bl = blockIdx.x * 256 + threadIdx.x;
    if (bl >= BL) return;
    const float* x = cat3 + (size_t)bl * (3 * NKc);
    float acc = ob[0];
    for (int k = 0; k < 3 * NKc; ++k) acc += x[k] * ow[k];
    out[bl] = acc;
}

// ================== host orchestration ==================
extern "C" void kernel_launch(void* const* d_in, const int* in_sizes, int n_in,
                              void* d_out, int out_size, void* d_ws, size_t ws_size,
                              hipStream_t stream)
{
    const float* context_x = (const float*)d_in[0];
    const float* value     = (const float*)d_in[1];
    const float* mask      = (const float*)d_in[2];
    const float* tmask     = (const float*)d_in[3];
    const float* edge_w    = (const float*)d_in[4];
    const float* edge_b    = (const float*)d_in[5];
    const float* chan_w    = (const float*)d_in[6];
    const float* chan_b    = (const float*)d_in[7];
    const float* time_w    = (const float*)d_in[8];
    const float* time_b    = (const float*)d_in[9];
    const float* corr      = (const float*)d_in[10];
    const float* out_w     = (const float*)d_in[11];
    const float* out_b     = (const float*)d_in[12];
    const float* ct_wq = (const float*)d_in[13]; const float* ct_bq = (const float*)d_in[14];
    const float* ct_wk = (const float*)d_in[15]; const float* ct_bk = (const float*)d_in[16];
    const float* ct_wv = (const float*)d_in[17]; const float* ct_bv = (const float*)d_in[18];
    const float* ct_wo = (const float*)d_in[19]; const float* ct_bo = (const float*)d_in[20];
    const float* tc_wq = (const float*)d_in[21]; const float* tc_bq = (const float*)d_in[22];
    const float* tc_wk = (const float*)d_in[23]; const float* tc_bk = (const float*)d_in[24];
    const float* tc_wv = (const float*)d_in[25]; const float* tc_bv = (const float*)d_in[26];
    const float* tc_wo = (const float*)d_in[27]; const float* tc_bo = (const float*)d_in[28];
    const float* en_w  = (const float*)d_in[29]; const float* en_b  = (const float*)d_in[30];
    const float* ca_wq = (const float*)d_in[31]; const float* ca_bq = (const float*)d_in[32];
    const float* ca_wk = (const float*)d_in[33]; const float* ca_bk = (const float*)d_in[34];
    const float* ca_wv = (const float*)d_in[35]; const float* ca_bv = (const float*)d_in[36];
    const float* ca_wo = (const float*)d_in[37]; const float* ca_bo = (const float*)d_in[38];
    const int*   i_patch = (const int*)d_in[39];
    (void)d_in[40]; (void)in_sizes; (void)n_in; (void)out_size; (void)ws_size;

    float* out = (float*)d_out;
    float* o_tgt   = out + (size_t)BL;
    float* o_src   = out + (size_t)2 * BL;
    float* o_smask = out + (size_t)3 * BL;
    float* o_cemb  = out + (size_t)4 * BL;

    // ----- workspace carve -----
    char* wsp = (char*)d_ws;
    auto alloc = [&](size_t bytes) -> void* {
        void* p = (void*)wsp;
        wsp += (bytes + 255) & ~(size_t)255;
        return p;
    };
    int*   T_inds = (int*)  alloc((size_t)BL * 4);
    int*   C_inds = (int*)  alloc((size_t)BL * 4);
    float* Uvals  = (float*)alloc((size_t)BL * 4);
    float* U      = (float*)alloc((size_t)BL * NKc * 4);
    float* TembA  = (float*)alloc((size_t)Bc * Tc * NKc * 4);
    float* TembB  = (float*)alloc((size_t)Bc * Tc * NKc * 4);
    float* Cemb   = (float*)alloc((size_t)Bc * Dc * NKc * 4);
    float* C2     = (float*)alloc((size_t)Bc * Dc * NKc * 4);
    float* Qp     = (float*)alloc((size_t)Bc * Tc * NKc * 4);
    float* AO     = (float*)alloc((size_t)Bc * Tc * NKc * 4);
    float* Kp     = (float*)alloc((size_t)BL * NKc * 4);
    float* Vp     = (float*)alloc((size_t)BL * NKc * 4);
    float* cat3   = (float*)alloc((size_t)BL * 3 * NKc * 4);   // cat2 aliases head of cat3
    float* cat2   = cat3;
    float* cpn    = (float*)alloc((size_t)Dc * Dc * 4);

    auto gemm = [&](const float* X, const float* W, const float* bias,
                    const float* R, float* Y, int M, int Kd, int mode) {
        dim3 g((unsigned)((M + 63) / 64), (unsigned)(NKc / 64));
        gemm_kernel<<<g, 256, 0, stream>>>(X, W, bias, R, Y, M, Kd, NKc, mode);
    };
    auto attn = [&](const float* Q, const float* K, const float* V,
                    const int* inds, float* AOp, int nq, int nk, int mm) {
        const int QT = (nq + 15) / 16;
        const int total = Bc * Hc * QT;
        attn_kernel<<<(total + 7) / 8, 256, 0, stream>>>(Q, K, V, inds, AOp, nq, nk, mm);
    };
    auto eblocks = [](long n) { return (unsigned)((n + 255) / 256); };

    // ----- stage 0: pack + embeddings -----
    pack_kernel<<<Bc, 256, 0, stream>>>(mask, tmask, value, T_inds, C_inds, Uvals,
                                        o_tgt, o_src, o_smask);
    corr_kernel<<<1, 256, 0, stream>>>(corr, i_patch, cpn);
    chan_embed_kernel<<<Dc, 128, 0, stream>>>(cpn, chan_w, chan_b, Cemb);
    time_embed_kernel<<<eblocks((long)Bc * Tc * NKc), 256, 0, stream>>>(
        context_x, time_w, time_b, TembA);
    edge_embed_kernel<<<eblocks((long)BL * NKc), 256, 0, stream>>>(
        Uvals, o_tgt, edge_w, edge_b, U);

    float* Temb = TembA;
    float* Tnxt = TembB;

    // ----- layers -----
    for (int i = 0; i < NLc; ++i) {
        const size_t oQ = (size_t)i * NKc * NKc;
        const size_t oK = (size_t)i * NKc * 2 * NKc;
        const size_t oE = (size_t)i * NKc * 3 * NKc;
        const size_t oB = (size_t)i * NKc;

        // --- channel <- (k_t, U) cross attention ---
        cat2_kernel<<<eblocks((long)BL * 2 * NKc), 256, 0, stream>>>(
            Temb, Tc, T_inds, U, cat2);
        gemm(Cemb, ct_wq + oQ, ct_bq + oB, nullptr, Qp, Bc * Dc, NKc, 0);
        gemm(cat2, ct_wk + oK, ct_bk + oB, nullptr, Kp, BL, 2 * NKc, 0);
        gemm(cat2, ct_wv + oK, ct_bv + oB, nullptr, Vp, BL, 2 * NKc, 0);
        attn(Qp, Kp, Vp, C_inds, AO, Dc, Lc, 1);
        gemm(AO, ct_wo + oQ, ct_bo + oB, AO, C2, Bc * Dc, NKc, 3);

        // --- time <- (k_c, U) cross attention ---
        cat2_kernel<<<eblocks((long)BL * 2 * NKc), 256, 0, stream>>>(
            Cemb, Dc, C_inds, U, cat2);
        gemm(Temb, tc_wq + oQ, tc_bq + oB, nullptr, Qp, Bc * Tc, NKc, 0);
        gemm(cat2, tc_wk + oK, tc_bk + oB, nullptr, Kp, BL, 2 * NKc, 0);
        gemm(cat2, tc_wv + oK, tc_bv + oB, nullptr, Vp, BL, 2 * NKc, 0);
        attn(Qp, Kp, Vp, T_inds, AO, Tc, Lc, 1);
        gemm(AO, tc_wo + oQ, tc_bo + oB, AO, Tnxt, Bc * Tc, NKc, 3);

        // --- edge feature update: U = relu(U + [U,k_t,k_c] @ en_w^T + en_b) ---
        cat3_kernel<<<eblocks((long)BL * 3 * NKc), 256, 0, stream>>>(
            U, Temb, T_inds, Cemb, C_inds, cat3);
        gemm(cat3, en_w + oE, en_b + oB, U, U, BL, 3 * NKc, 2);

        // --- channel self attention: C_emb = MAB2(C2, C2) ---
        gemm(C2, ca_wq + oQ, ca_bq + oB, nullptr, Qp, Bc * Dc, NKc, 0);
        gemm(C2, ca_wk + oQ, ca_bk + oB, nullptr, Kp, Bc * Dc, NKc, 0);
        gemm(C2, ca_wv + oQ, ca_bv + oB, nullptr, Vp, Bc * Dc, NKc, 0);
        attn(Qp, Kp, Vp, nullptr, AO, Dc, Dc, 0);
        gemm(AO, ca_wo + oQ, ca_bo + oB, AO, Cemb, Bc * Dc, NKc, 3);

        // T_emb = T2 (pointer swap)
        float* tmp = Temb; Temb = Tnxt; Tnxt = tmp;
    }

    // ----- final projection + outputs -----
    cat3_kernel<<<eblocks((long)BL * 3 * NKc), 256, 0, stream>>>(
        U, Temb, T_inds, Cemb, C_inds, cat3);
    outproj_kernel<<<eblocks((long)BL), 256, 0, stream>>>(cat3, out_w, out_b, out);
    hipMemcpyAsync(o_cemb, Cemb, (size_t)Bc * Dc * NKc * 4,
                   hipMemcpyDeviceToDevice, stream);
}
